// CosineDistance_15685220565422
// MI455X (gfx1250) — 8-bit matrix operand order repaired and checked by reading, compiled, not run on hardware
//
#include <hip/hip_runtime.h>
#include <hip/hip_bf16.h>

typedef int v8i __attribute__((ext_vector_type(8)));

#define D        1024
#define BX       4096
#define MW       8192
#define QMAX     127.0f

#define OFF_MAXX   0u
#define OFF_MAXW   4u
#define OFF_NORMX  16u
#define OFF_NORMW  (OFF_NORMX + BX * 4u)
#define OFF_QX     (OFF_NORMW + MW * 4u)
#define OFF_QW     (OFF_QX + (unsigned)(BX * D))

__global__ void init_scales(unsigned* __restrict__ mx, unsigned* __restrict__ mw) {
    if (threadIdx.x == 0) { *mx = 0u; *mw = 0u; }
}

__global__ __launch_bounds__(256)
void rownorm_rowmax(const float* __restrict__ a,
                    float* __restrict__ norms,
                    unsigned* __restrict__ maxbits) {
    __shared__ float s[256];
    const int tid = threadIdx.x;
    const float* ar = a + (size_t)blockIdx.x * D;

    float v0 = ar[tid];
    float v1 = ar[tid + 256];
    float v2 = ar[tid + 512];
    float v3 = ar[tid + 768];

    s[tid] = v0 * v0 + v1 * v1 + v2 * v2 + v3 * v3;
    __syncthreads();
    #pragma unroll
    for (int off = 128; off > 0; off >>= 1) {
        if (tid < off) s[tid] += s[tid + off];
        __syncthreads();
    }
    const float denom = fmaxf(sqrtf(s[0]), 1e-12f);
    __syncthreads();
    if (tid == 0) norms[blockIdx.x] = denom;

    float m = fmaxf(fmaxf(fabsf(v0), fabsf(v1)), fmaxf(fabsf(v2), fabsf(v3))) / denom;
    s[tid] = m;
    __syncthreads();
    #pragma unroll
    for (int off = 128; off > 0; off >>= 1) {
        if (tid < off) s[tid] = fmaxf(s[tid], s[tid + off]);
        __syncthreads();
    }
    if (tid == 0) atomicMax(maxbits, __float_as_uint(s[0]));
}

__global__ __launch_bounds__(256)
void quantize_i8(const float* __restrict__ a,
                 const float* __restrict__ norms,
                 const unsigned* __restrict__ maxbits,
                 char* __restrict__ q) {
    const size_t i4  = ((size_t)blockIdx.x * blockDim.x + threadIdx.x) * 4;
    const int    row = (int)(i4 >> 10);
    const float  scale = __uint_as_float(*maxbits);
    const float  inv   = QMAX / (scale * norms[row]);

    const float4 v = *reinterpret_cast<const float4*>(a + i4);
    const int b0 = (int)rintf(v.x * inv);
    const int b1 = (int)rintf(v.y * inv);
    const int b2 = (int)rintf(v.z * inv);
    const int b3 = (int)rintf(v.w * inv);
    const int packed = (b0 & 0xff) | ((b1 & 0xff) << 8) |
                       ((b2 & 0xff) << 16) | (b3 << 24);
    *reinterpret_cast<int*>(q + i4) = packed;
}

__global__ __launch_bounds__(256)
void gemm_i8_wmma(const char* __restrict__ qx,
                  const char* __restrict__ qw,
                  const unsigned* __restrict__ scales,
                  float* __restrict__ out) {
    const int tid  = threadIdx.x;
    const int wave = tid >> 5;
    const int lane = tid & 31;
    const int lrow = lane & 15;
    const int half = lane >> 4;

    const int x0 = blockIdx.y * 64 + (wave >> 2) * 32;
    const int n0 = blockIdx.x * 256 + (wave & 3) * 64;

    const char* abase = qx + (size_t)(x0 + lrow) * D + half * 8;
    const char* bbase = qw + (size_t)(n0 + lrow) * D + half * 8;

    v8i acc[2][4];
    #pragma unroll
    for (int u = 0; u < 2; ++u)
        #pragma unroll
        for (int t = 0; t < 4; ++t)
            acc[u][t] = (v8i){0, 0, 0, 0, 0, 0, 0, 0};

    #pragma unroll
    for (int k0 = 0; k0 < D; k0 += 64) {
        v8i a[2];
        #pragma unroll
        for (int u = 0; u < 2; ++u)
            #pragma unroll
            for (int p = 0; p < 4; ++p) {
                const uint2 t2 = *reinterpret_cast<const uint2*>(
                    abase + u * 16 * D + k0 + p * 16);
                a[u][2 * p]     = (int)t2.x;
                a[u][2 * p + 1] = (int)t2.y;
            }

        v8i b[4];
        #pragma unroll
        for (int t = 0; t < 4; ++t) {
            #pragma unroll
            for (int p = 0; p < 4; ++p) {
                const uint2 t2 = *reinterpret_cast<const uint2*>(
                    bbase + t * 16 * D + k0 + p * 16);
                b[t][2 * p]     = (int)t2.x;
                b[t][2 * p + 1] = (int)t2.y;
            }
        }

        #pragma unroll
        for (int t = 0; t < 4; ++t)
            #pragma unroll
            for (int u = 0; u < 2; ++u)
                acc[u][t] = __builtin_amdgcn_wmma_i32_16x16x64_iu8(
                     true, a[u],  true, b[t], acc[u][t],
                     false,  false);
    }

    const float sx = __uint_as_float(scales[0]);
    const float sw = __uint_as_float(scales[1]);
    const float factor = (sx * sw) * (1.0f / (QMAX * QMAX));

    float* obase = out + (size_t)(x0 + 8 * half) * MW + n0 + lrow;
    #pragma unroll
    for (int u = 0; u < 2; ++u)
        #pragma unroll
        for (int t = 0; t < 4; ++t)
            #pragma unroll
            for (int r = 0; r < 8; ++r)
                obase[(size_t)(u * 16 + r) * MW + t * 16] =
                    1.0f - (float)acc[u][t][r] * factor;
}

extern "C" void kernel_launch(void* const* d_in, const int* in_sizes, int n_in,
                              void* d_out, int out_size, void* d_ws, size_t ws_size,
                              hipStream_t stream) {
    const float* x = (const float*)d_in[0];
    const float* w = (const float*)d_in[1];
    float* out = (float*)d_out;

    char* ws = (char*)d_ws;
    unsigned* maxx  = (unsigned*)(ws + OFF_MAXX);
    unsigned* maxw  = (unsigned*)(ws + OFF_MAXW);
    float*    normx = (float*)(ws + OFF_NORMX);
    float*    normw = (float*)(ws + OFF_NORMW);
    char*     qx    = ws + OFF_QX;
    char*     qw    = ws + OFF_QW;

    init_scales<<<1, 64, 0, stream>>>(maxx, maxw);

    rownorm_rowmax<<<BX, 256, 0, stream>>>(x, normx, maxx);
    rownorm_rowmax<<<MW, 256, 0, stream>>>(w, normw, maxw);

    quantize_i8<<<(BX * D) / (256 * 4), 256, 0, stream>>>(x, normx, maxx, qx);
    quantize_i8<<<(MW * D) / (256 * 4), 256, 0, stream>>>(w, normw, maxw, qw);

    dim3 grid(MW / 256, BX / 64);
    gemm_i8_wmma<<<grid, 256, 0, stream>>>(qx, qw, (const unsigned*)ws, out);
}
